// MultiHeadSelfAttentionLayer_9552007266773
// MI455X (gfx1250) — matrix operand order repaired and checked by reading, not compiled, not run on hardware
//
#include <hip/hip_runtime.h>
#include <math.h>

// ---------------------------------------------------------------------------
// Transformer encoder layer for MI455X (gfx1250, wave32, WMMA + TDM).
// GEMMs: v_wmma_f32_16x16x32_f16 fed from LDS tiles staged by the Tensor
// Data Mover (tensor_load_to_lds, double-buffered, s_wait_tensorcnt).
// Attention: flash-style online softmax, never materializing S x S.
// ---------------------------------------------------------------------------

#define DEVINL __device__ __forceinline__

typedef __attribute__((ext_vector_type(16))) _Float16 v16h;
typedef __attribute__((ext_vector_type(8)))  _Float16 v8h;
typedef __attribute__((ext_vector_type(8)))  float    v8f;
typedef __attribute__((ext_vector_type(4)))  unsigned int v4u;
typedef __attribute__((ext_vector_type(8)))  int      v8i;
typedef __attribute__((ext_vector_type(4)))  int      v4i;

static constexpr int Bn     = 8;
static constexpr int Sn     = 2048;
static constexpr int Hn     = 512;
static constexpr int HEADSn = 8;
static constexpr int DHn    = 64;
static constexpr int Mn     = Bn * Sn;   // 16384 rows
static constexpr int FFn    = 4 * Hn;    // 2048

#if defined(__has_builtin)
#if __has_builtin(__builtin_amdgcn_tensor_load_to_lds) && __has_builtin(__builtin_amdgcn_s_wait_tensorcnt)
#define USE_TDM 1
#endif
#endif
#ifndef USE_TDM
#define USE_TDM 0
#endif

union FragU { v16h v; v8h h[2]; };

// A operand, 16x32 f16 (MxK). Lane m (0-15): K 0..15 contiguous; lane m+16: K 16..31 contiguous.
DEVINL v16h load_a_frag(const _Float16* base, int stride) {
  const int lane = threadIdx.x & 31;
  const int r    = lane & 15;
  const int kb   = (lane & 16) ? 16 : 0;
  const _Float16* p = base + (size_t)r * stride + kb;
  FragU f;
  f.h[0] = *(const v8h*)(p);        // K = kb .. kb+7
  f.h[1] = *(const v8h*)(p + 8);    // K = kb+8 .. kb+15
  return f.v;
}

// B operand, 32x16 f16 (KxN), sourced from an [N, K]-major matrix (i.e. B^T).
// Lane n (0-15): K 0..15 contiguous; lane n+16: K 16..31 contiguous.
DEVINL v16h load_b_frag(const _Float16* base, int stride) {
  const int lane = threadIdx.x & 31;
  const _Float16* p = base + (size_t)(lane & 15) * stride + ((lane & 16) ? 16 : 0);
  return *(const v16h*)(p);
}

DEVINL v8f wmma_f16(v16h a, v16h b, v8f c) {
  return __builtin_amdgcn_wmma_f32_16x16x32_f16(false, a, false, b, (short)0, c,
                                                false, false);
}

#if USE_TDM
// Issue a 2D TDM tile load (f16 elements): global [numRows x rowLen] row-major,
// tile tileH x tileW at gptr, packed into LDS at ldsOff with row stride tileW.
// D# layout per CDNA5 ISA.
DEVINL void tdm_load_2d(unsigned ldsOff, const _Float16* gptr,
                        int rowLen, int numRows, int tileW, int tileH,
                        int strideElems) {
  const unsigned long long ga = (unsigned long long)gptr;
  v4u g0;
  g0[0] = 1u;                                   // count=1, user descriptor
  g0[1] = ldsOff;                               // lds_addr (bytes)
  g0[2] = (unsigned)(ga & 0xffffffffull);       // global_addr[31:0]
  g0[3] = (unsigned)((ga >> 32) & 0x01ffffffull) | (2u << 30);  // addr[56:32], type=2
  v8i g1;
  g1[0] = 1 << 16;                                              // data_size=1 (2B)
  g1[1] = (int)((unsigned)(rowLen & 0xffff) << 16);             // tensor_dim0[15:0]
  g1[2] = (int)(((unsigned)(rowLen >> 16) & 0xffffu) |
                ((unsigned)(numRows & 0xffff) << 16));          // dim0 hi | dim1 lo
  g1[3] = (int)(((unsigned)(numRows >> 16) & 0xffffu) |
                ((unsigned)(tileW & 0xffff) << 16));            // dim1 hi | tile_dim0
  g1[4] = (int)((unsigned)(tileH & 0xffff));                    // tile_dim1 (tile_dim2=0)
  g1[5] = strideElems;                                          // dim0_stride[31:0]
  g1[6] = 0;                                                    // stride hi | dim1_stride lo
  g1[7] = 0;
  const v4i z4 = {0, 0, 0, 0};
#if __clang_major__ >= 23
  const v8i z8 = {0, 0, 0, 0, 0, 0, 0, 0};
  __builtin_amdgcn_tensor_load_to_lds(g0, g1, z4, z4, z8, 0);
#else
  __builtin_amdgcn_tensor_load_to_lds(g0, g1, z4, z4, 0);
#endif
}
#endif

// ---------------------------------------------------------------------------
// Conversion helpers
// ---------------------------------------------------------------------------

__global__ void f32_to_f16_kernel(const float* __restrict__ in,
                                  _Float16* __restrict__ out, size_t n) {
  size_t i = (size_t)blockIdx.x * blockDim.x + threadIdx.x;
  if (i < n) out[i] = (_Float16)in[i];
}

// in: f32 [K, N] row-major  ->  out: f16 [N, K] (transposed)
__global__ void transpose_to_f16_kernel(const float* __restrict__ in,
                                        _Float16* __restrict__ out,
                                        int K, int N) {
  size_t idx = (size_t)blockIdx.x * blockDim.x + threadIdx.x;
  if (idx >= (size_t)K * N) return;
  int n = (int)(idx / K);
  int k = (int)(idx % K);
  out[idx] = (_Float16)in[(size_t)k * N + n];
}

// ---------------------------------------------------------------------------
// WMMA GEMM: C[M,N] = A[M,K] (f16, row-major) * Bt[N,K]^T + bias
// Block: 128 threads = 4 waves. Block tile 128x64; wave tile 32x64.
// TDM double-buffers A (128x32) and B (64x32) k-slices in LDS.
// ---------------------------------------------------------------------------

enum StoreMode { ST_F16 = 0, ST_F16_VT = 1, ST_F16_RELU = 2, ST_F32_RES = 3 };

template <int MODE>
__global__ __launch_bounds__(128)
void wmma_gemm_kernel(const _Float16* __restrict__ A,
                      const _Float16* __restrict__ Bt,
                      const float* __restrict__ bias,
                      const float* __restrict__ resid,
                      float* __restrict__ outF,
                      _Float16* __restrict__ outH,
                      int Mdim, int Ndim, int Kdim) {
  const int lane = threadIdx.x & 31;
  const int wave = threadIdx.x >> 5;
  const int m0 = blockIdx.x * 128 + wave * 32;  // wave's 32 rows
  const int n0 = blockIdx.y * 64;

  v8f acc[2][4] = {};

#if USE_TDM
  __shared__ __align__(16) _Float16 ldsA[2][128 * 32];
  __shared__ __align__(16) _Float16 ldsB[2][64 * 32];
  const int blockM = blockIdx.x * 128;
  const int nk = Kdim / 32;

  if (wave == 0) {
    tdm_load_2d((unsigned)(size_t)&ldsA[0][0], A + (size_t)blockM * Kdim,
                Kdim, Mdim, 32, 128, Kdim);
    tdm_load_2d((unsigned)(size_t)&ldsB[0][0], Bt + (size_t)n0 * Kdim,
                Kdim, Ndim, 32, 64, Kdim);
  }
  for (int ik = 0; ik < nk; ++ik) {
    const int buf = ik & 1;
    if (wave == 0) {
      if (ik + 1 < nk) {
        const int k = (ik + 1) * 32;
        tdm_load_2d((unsigned)(size_t)&ldsA[buf ^ 1][0],
                    A + (size_t)blockM * Kdim + k, Kdim, Mdim, 32, 128, Kdim);
        tdm_load_2d((unsigned)(size_t)&ldsB[buf ^ 1][0],
                    Bt + (size_t)n0 * Kdim + k, Kdim, Ndim, 32, 64, Kdim);
        __builtin_amdgcn_s_wait_tensorcnt((short)2);  // current buffers landed
      } else {
        __builtin_amdgcn_s_wait_tensorcnt((short)0);
      }
    }
    __syncthreads();

    const _Float16* Aw = &ldsA[buf][wave * 32 * 32];
    const v16h a0 = load_a_frag(Aw, 32);
    const v16h a1 = load_a_frag(Aw + 16 * 32, 32);
#pragma unroll
    for (int t = 0; t < 4; ++t) {
      const v16h b = load_b_frag(&ldsB[buf][t * 16 * 32], 32);
      acc[0][t] = wmma_f16(a0, b, acc[0][t]);
      acc[1][t] = wmma_f16(a1, b, acc[1][t]);
    }
    __syncthreads();  // safe to overwrite this buffer next round
  }
#else
  for (int k = 0; k < Kdim; k += 32) {
    const v16h a0 = load_a_frag(A + (size_t)m0 * Kdim + k, Kdim);
    const v16h a1 = load_a_frag(A + (size_t)(m0 + 16) * Kdim + k, Kdim);
#pragma unroll
    for (int t = 0; t < 4; ++t) {
      const v16h b = load_b_frag(Bt + (size_t)(n0 + 16 * t) * Kdim + k, Kdim);
      acc[0][t] = wmma_f16(a0, b, acc[0][t]);
      acc[1][t] = wmma_f16(a1, b, acc[1][t]);
    }
  }
#endif

  // C/D layout: acc[r] on lanes 0-15 -> row r, lanes 16-31 -> row r+8; col = lane&15
  const int hi = (lane & 16) ? 8 : 0;
  const int cn = lane & 15;
#pragma unroll
  for (int sub = 0; sub < 2; ++sub) {
#pragma unroll
    for (int t = 0; t < 4; ++t) {
      const int n = n0 + 16 * t + cn;
      const float bv = bias[n];
#pragma unroll
      for (int r = 0; r < 8; ++r) {
        const int m = m0 + sub * 16 + r + hi;
        float v = acc[sub][t][r] + bv;
        if (MODE == ST_F16_RELU) v = fmaxf(v, 0.0f);
        if (MODE == ST_F32_RES) {
          outF[(size_t)m * Ndim + n] = v + resid[(size_t)m * Ndim + n];
        } else if (MODE == ST_F16_VT) {
          // n = h*DH + d, m = b*S + s -> Vt[((b*HEADS + h)*DH + d)*S + s]
          const int bb = m / Sn, s = m % Sn;
          const int h = n / DHn, d = n % DHn;
          outH[(((size_t)bb * HEADSn + h) * DHn + d) * Sn + s] = (_Float16)v;
        } else {
          outH[(size_t)m * Ndim + n] = (_Float16)v;
        }
      }
    }
  }
}

// ---------------------------------------------------------------------------
// Flash attention: one wave per (b, h, 16 query rows). DH = 64.
// Qh/Kh: f16 [B,S,HEADS*DH] row-major. Vt: f16 [B,HEADS,DH,S]. Oh like Qh.
// ---------------------------------------------------------------------------

__global__ __launch_bounds__(128)
void flash_attn_kernel(const _Float16* __restrict__ Qh,
                       const _Float16* __restrict__ Kh,
                       const _Float16* __restrict__ Vt,
                       const unsigned char* __restrict__ mask,
                       _Float16* __restrict__ Oh) {
  __shared__ __align__(32) _Float16 ldsP[4][16 * 32];  // per-wave P tile (16x32)

  const int lane = threadIdx.x & 31;
  const int wave = threadIdx.x >> 5;
  const int gw = blockIdx.x * 4 + wave;     // 8192 waves total
  const int ib = gw & 127;                  // S/16 = 128
  const int h  = (gw >> 7) & 7;
  const int b  = gw >> 10;
  const int i0 = ib * 16;

  const _Float16* Qbase = Qh + (size_t)(b * Sn + i0) * Hn + h * DHn;
  const _Float16* Kbase = Kh + (size_t)b * Sn * Hn + h * DHn;
  const _Float16* Vbase = Vt + (size_t)(b * HEADSn + h) * DHn * Sn;

  const v16h qf0 = load_a_frag(Qbase, Hn);        // d = 0..31
  const v16h qf1 = load_a_frag(Qbase + 32, Hn);   // d = 32..63

  v8f o[4] = {};
  float rm[8], rs[8];
#pragma unroll
  for (int r = 0; r < 8; ++r) { rm[r] = -__builtin_inff(); rs[r] = 0.0f; }

  const int hi = (lane & 16) ? 8 : 0;
  const int cn = lane & 15;
  const float scale = 0.125f;  // 1/sqrt(64)
  const bool hiHalf = (lane & 16) != 0;

  _Float16* P = ldsP[wave];

  for (int j0 = 0; j0 < Sn; j0 += 32) {
    // ---- scores for two 16-wide j tiles: s0 (j0..j0+15), s1 (j0+16..j0+31)
    v8f s0 = {}, s1 = {};
    {
      v16h k00 = load_b_frag(Kbase + (size_t)j0 * Hn, Hn);
      v16h k01 = load_b_frag(Kbase + (size_t)j0 * Hn + 32, Hn);
      s0 = wmma_f16(qf0, k00, s0);
      s0 = wmma_f16(qf1, k01, s0);
      v16h k10 = load_b_frag(Kbase + (size_t)(j0 + 16) * Hn, Hn);
      v16h k11 = load_b_frag(Kbase + (size_t)(j0 + 16) * Hn + 32, Hn);
      s1 = wmma_f16(qf0, k10, s1);
      s1 = wmma_f16(qf1, k11, s1);
    }

    // ---- scale + mask + new row max (reduce across 16 lanes of this half).
    // Mask: lane reads mask[row, j0+lane] (coalesced 32B/row); shfl_xor(16)
    // swaps the two 16-wide halves so each lane gets cols cn and cn+16.
    float mnew[8];
#pragma unroll
    for (int r = 0; r < 8; ++r) {
      const int i = i0 + r + hi;
      const size_t mrow = ((size_t)b * Sn + i) * Sn + j0;
      const int mv = (int)mask[mrow + lane];
      const int mw = __shfl_xor(mv, 16, 32);
      const int mt0 = hiHalf ? mw : mv;   // col cn      (tile 0)
      const int mt1 = hiHalf ? mv : mw;   // col cn + 16 (tile 1)
      float a = s0[r] * scale;
      float c = s1[r] * scale;
      if (mt0) a = -__builtin_inff();
      if (mt1) c = -__builtin_inff();
      s0[r] = a;
      s1[r] = c;
      float mx = fmaxf(a, c);
      mx = fmaxf(mx, __shfl_xor(mx, 1, 16));
      mx = fmaxf(mx, __shfl_xor(mx, 2, 16));
      mx = fmaxf(mx, __shfl_xor(mx, 4, 16));
      mx = fmaxf(mx, __shfl_xor(mx, 8, 16));
      mnew[r] = fmaxf(rm[r], mx);
    }

    // ---- exponentiate, stage P in LDS, update running stats, rescale output
#pragma unroll
    for (int r = 0; r < 8; ++r) {
      const float e0 = __expf(s0[r] - mnew[r]);
      const float e1 = __expf(s1[r] - mnew[r]);
      const int row = r + hi;
      P[row * 32 + cn]      = (_Float16)e0;
      P[row * 32 + 16 + cn] = (_Float16)e1;
      float t = e0 + e1;
      t += __shfl_xor(t, 1, 16);
      t += __shfl_xor(t, 2, 16);
      t += __shfl_xor(t, 4, 16);
      t += __shfl_xor(t, 8, 16);
      const float corr = __expf(rm[r] - mnew[r]);
      rs[r] = rs[r] * corr + t;
      rm[r] = mnew[r];
#pragma unroll
      for (int d = 0; d < 4; ++d) o[d][r] *= corr;
    }

    // wave-private LDS tile: wait for all lanes' DS stores before re-reading
    asm volatile("s_wait_dscnt 0" ::: "memory");

    // ---- P(16x32) x V(32x64): 4 WMMAs against transposed V
    const v16h pf = load_a_frag(P, 32);
#pragma unroll
    for (int d = 0; d < 4; ++d) {
      v16h vf = load_b_frag(Vbase + (size_t)(d * 16) * Sn + j0, Sn);
      o[d] = wmma_f16(pf, vf, o[d]);
    }
  }

  // ---- normalize and store attention output (f16, row-major [M, H])
#pragma unroll
  for (int d = 0; d < 4; ++d) {
#pragma unroll
    for (int r = 0; r < 8; ++r) {
      const float v = o[d][r] / rs[r];
      const size_t m = (size_t)b * Sn + i0 + r + hi;
      Oh[m * Hn + h * DHn + d * 16 + cn] = (_Float16)v;
    }
  }
}

// ---------------------------------------------------------------------------
// LayerNorm over H=512; one wave per row (8 rows per 256-thread block).
// ---------------------------------------------------------------------------

__global__ __launch_bounds__(256)
void layernorm_kernel(const float* __restrict__ x,
                      const float* __restrict__ g,
                      const float* __restrict__ be,
                      float* __restrict__ outF,
                      _Float16* __restrict__ outH, int rows) {
  const int lane = threadIdx.x & 31;
  const int wave = threadIdx.x >> 5;
  const int row = blockIdx.x * 8 + wave;
  if (row >= rows) return;
  const float* p = x + (size_t)row * Hn;

  float vals[16];
  float s = 0.0f;
#pragma unroll
  for (int i = 0; i < 16; ++i) {
    vals[i] = p[i * 32 + lane];
    s += vals[i];
  }
  s += __shfl_xor(s, 1);  s += __shfl_xor(s, 2);
  s += __shfl_xor(s, 4);  s += __shfl_xor(s, 8);  s += __shfl_xor(s, 16);
  const float mean = s * (1.0f / Hn);

  float vsum = 0.0f;
#pragma unroll
  for (int i = 0; i < 16; ++i) {
    const float d = vals[i] - mean;
    vsum += d * d;
  }
  vsum += __shfl_xor(vsum, 1);  vsum += __shfl_xor(vsum, 2);
  vsum += __shfl_xor(vsum, 4);  vsum += __shfl_xor(vsum, 8);
  vsum += __shfl_xor(vsum, 16);
  const float inv = rsqrtf(vsum * (1.0f / Hn) + 1e-5f);

#pragma unroll
  for (int i = 0; i < 16; ++i) {
    const int c = i * 32 + lane;
    const float y = (vals[i] - mean) * inv * g[c] + be[c];
    outF[(size_t)row * Hn + c] = y;
    if (outH) outH[(size_t)row * Hn + c] = (_Float16)y;
  }
}

// ---------------------------------------------------------------------------
// Host-side orchestration
// ---------------------------------------------------------------------------

extern "C" void kernel_launch(void* const* d_in, const int* in_sizes, int n_in,
                              void* d_out, int out_size, void* d_ws, size_t ws_size,
                              hipStream_t stream) {
  (void)in_sizes; (void)n_in; (void)out_size; (void)ws_size;

  const float* X    = (const float*)d_in[0];
  const unsigned char* mask = (const unsigned char*)d_in[1];
  const float* Wq = (const float*)d_in[2];  const float* bq  = (const float*)d_in[3];
  const float* Wk = (const float*)d_in[4];  const float* bk  = (const float*)d_in[5];
  const float* Wv = (const float*)d_in[6];  const float* bv  = (const float*)d_in[7];
  const float* Wo = (const float*)d_in[8];  const float* bo  = (const float*)d_in[9];
  const float* g1 = (const float*)d_in[10]; const float* b1  = (const float*)d_in[11];
  const float* W1 = (const float*)d_in[12]; const float* bf1 = (const float*)d_in[13];
  const float* W2 = (const float*)d_in[14]; const float* bf2 = (const float*)d_in[15];
  const float* g2 = (const float*)d_in[16]; const float* b2  = (const float*)d_in[17];
  float* out = (float*)d_out;

  // --- workspace carve-out (256-byte aligned)
  size_t off = 0;
  auto alloc = [&](size_t bytes) -> void* {
    void* p = (char*)d_ws + off;
    off += (bytes + 255) & ~(size_t)255;
    return p;
  };
  const size_t MH = (size_t)Mn * Hn;
  _Float16* Xh    = (_Float16*)alloc(MH * 2);
  _Float16* WqT   = (_Float16*)alloc((size_t)Hn * Hn * 2);
  _Float16* WkT   = (_Float16*)alloc((size_t)Hn * Hn * 2);
  _Float16* WvT   = (_Float16*)alloc((size_t)Hn * Hn * 2);
  _Float16* WoT   = (_Float16*)alloc((size_t)Hn * Hn * 2);
  _Float16* W1T   = (_Float16*)alloc((size_t)Hn * FFn * 2);
  _Float16* W2T   = (_Float16*)alloc((size_t)FFn * Hn * 2);
  _Float16* Qh    = (_Float16*)alloc(MH * 2);
  _Float16* Kh    = (_Float16*)alloc(MH * 2);
  _Float16* Vth   = (_Float16*)alloc(MH * 2);
  _Float16* attnh = (_Float16*)alloc(MH * 2);
  float*    y1    = (float*)alloc(MH * 4);
  float*    ln1   = (float*)alloc(MH * 4);
  _Float16* ln1h  = (_Float16*)alloc(MH * 2);
  _Float16* ffh   = (_Float16*)alloc((size_t)Mn * FFn * 2);
  float*    y2    = (float*)alloc(MH * 4);

  // --- precision conversion (f32 -> f16; weights transposed to [N,K])
  f32_to_f16_kernel<<<(unsigned)((MH + 255) / 256), 256, 0, stream>>>(X, Xh, MH);
  {
    const size_t whh = (size_t)Hn * Hn;
    transpose_to_f16_kernel<<<(unsigned)((whh + 255) / 256), 256, 0, stream>>>(Wq, WqT, Hn, Hn);
    transpose_to_f16_kernel<<<(unsigned)((whh + 255) / 256), 256, 0, stream>>>(Wk, WkT, Hn, Hn);
    transpose_to_f16_kernel<<<(unsigned)((whh + 255) / 256), 256, 0, stream>>>(Wv, WvT, Hn, Hn);
    transpose_to_f16_kernel<<<(unsigned)((whh + 255) / 256), 256, 0, stream>>>(Wo, WoT, Hn, Hn);
    const size_t wff = (size_t)Hn * FFn;
    transpose_to_f16_kernel<<<(unsigned)((wff + 255) / 256), 256, 0, stream>>>(W1, W1T, Hn, FFn);
    transpose_to_f16_kernel<<<(unsigned)((wff + 255) / 256), 256, 0, stream>>>(W2, W2T, FFn, Hn);
  }

  const dim3 blk(128);
  const dim3 gH(Mn / 128, Hn / 64);    // (128, 8)
  const dim3 gF(Mn / 128, FFn / 64);   // (128, 32)

  // --- QKV projections
  wmma_gemm_kernel<ST_F16><<<gH, blk, 0, stream>>>(Xh, WqT, bq, nullptr, nullptr, Qh, Mn, Hn, Hn);
  wmma_gemm_kernel<ST_F16><<<gH, blk, 0, stream>>>(Xh, WkT, bk, nullptr, nullptr, Kh, Mn, Hn, Hn);
  wmma_gemm_kernel<ST_F16_VT><<<gH, blk, 0, stream>>>(Xh, WvT, bv, nullptr, nullptr, Vth, Mn, Hn, Hn);

  // --- attention (8192 waves = 2048 blocks x 4 waves)
  flash_attn_kernel<<<2048, blk, 0, stream>>>(Qh, Kh, Vth, mask, attnh);

  // --- output projection + residual, then LN1
  wmma_gemm_kernel<ST_F32_RES><<<gH, blk, 0, stream>>>(attnh, WoT, bo, X, y1, nullptr, Mn, Hn, Hn);
  layernorm_kernel<<<Mn / 8, 256, 0, stream>>>(y1, g1, b1, ln1, ln1h, Mn);

  // --- FFN
  wmma_gemm_kernel<ST_F16_RELU><<<gF, blk, 0, stream>>>(ln1h, W1T, bf1, nullptr, nullptr, ffh, Mn, FFn, Hn);
  wmma_gemm_kernel<ST_F32_RES><<<gH, blk, 0, stream>>>(ffh, W2T, bf2, ln1, y2, nullptr, Mn, Hn, FFn);

  // --- LN2 -> final f32 output
  layernorm_kernel<<<Mn / 8, 256, 0, stream>>>(y2, g2, b2, out, nullptr, Mn);
}
